// Head_31885837205564
// MI455X (gfx1250) — matrix operand order repaired and checked by reading, not compiled, not run on hardware
//
#include <hip/hip_runtime.h>

typedef __attribute__((ext_vector_type(16))) _Float16 v16h;
typedef __attribute__((ext_vector_type(4)))  _Float16 v4h;
typedef __attribute__((ext_vector_type(8)))  float    v8f;
typedef __attribute__((ext_vector_type(4)))  float    v4f;

#define WMMA_F16(A, B, C) \
  __builtin_amdgcn_wmma_f32_16x16x32_f16(false, (A), false, (B), (short)0, (C), false, false)

constexpr int Bn = 8, Tn = 2048, En = 1024, Hn = 64;

// ---------------------------------------------------------------------------
// Kernel 0: one-shot fp32 -> f16 conversion (used for Wq/Wk/Wv).
// ---------------------------------------------------------------------------
__global__ __launch_bounds__(256) void cvt_f32_f16_kernel(
    const float* __restrict__ src, _Float16* __restrict__ dst, int n4)
{
  const int i = blockIdx.x * blockDim.x + threadIdx.x;
  if (i < n4) {
    const v4f f = *(const v4f*)(src + 4 * (size_t)i);
    v4h h;
#pragma unroll
    for (int j = 0; j < 4; ++j) h[j] = (_Float16)f[j];
    *(v4h*)(dst + 4 * (size_t)i) = h;
  }
}

// ---------------------------------------------------------------------------
// Kernel 1: QKV projection.  One wave per 16-token tile (1024 waves).
// WMMA f32_16x16x32_f16; A = x tile (fp32 read once, converted in-loop),
// B = W^T tile (pre-converted f16 -> pure b128 vector loads).
// ---------------------------------------------------------------------------
__global__ __launch_bounds__(256) void qkv_kernel(
    const float* __restrict__ x,
    const _Float16* __restrict__ Whq, const float* __restrict__ bq,
    const _Float16* __restrict__ Whk, const float* __restrict__ bk,
    const _Float16* __restrict__ Whv, const float* __restrict__ bv,
    _Float16* __restrict__ Qh, _Float16* __restrict__ Kh, _Float16* __restrict__ Vh)
{
  const int lane = threadIdx.x & 31;
  const int wave = (blockIdx.x * blockDim.x + threadIdx.x) >> 5;  // 0..1023
  const int bIdx = wave >> 7;
  const int t0   = (wave & 127) << 4;
  const int nl   = lane & 15;
  const int hl   = lane >> 4;
  const int kb8  = hl * 8;
  const int kb16 = hl * 16;

  const float* xr = x + ((size_t)bIdx * Tn + t0 + nl) * En;

  v8f acc[3][4] = {};  // {Q,K,V} x 4 column tiles of 16

  const _Float16* Ws[3] = {Whq, Whk, Whv};

  for (int kk = 0; kk < En; kk += 32) {
    v16h a;
#pragma unroll
    for (int j = 0; j < 8; ++j) {
      a[j]     = (_Float16)xr[kk + kb16 + j];
      a[8 + j] = (_Float16)xr[kk + kb16 + 8 + j];
    }
#pragma unroll
    for (int m = 0; m < 3; ++m) {
#pragma unroll
      for (int nt = 0; nt < 4; ++nt) {
        const v16h bb =
            *(const v16h*)(Ws[m] + (size_t)(nt * 16 + nl) * En + kk + kb16);
        acc[m][nt] = WMMA_F16(a, bb, acc[m][nt]);
      }
    }
  }

  const float* bs[3]  = {bq, bk, bv};
  _Float16*   outs[3] = {Qh, Kh, Vh};
#pragma unroll
  for (int m = 0; m < 3; ++m) {
#pragma unroll
    for (int nt = 0; nt < 4; ++nt) {
      const float bias = bs[m][nt * 16 + nl];
#pragma unroll
      for (int r = 0; r < 8; ++r) {
        outs[m][((size_t)bIdx * Tn + t0 + r + 8 * hl) * Hn + nt * 16 + nl] =
            (_Float16)(acc[m][nt][r] + bias);
      }
    }
  }
}

// ---------------------------------------------------------------------------
// Kernel 2: flash attention.  One wave per (batch, 16-query tile).
// 64-key blocks: 4 S-tiles per iteration to amortize the softmax VALU
// (one shuffle-reduce tree + one running-stat update per 64 keys).
// ---------------------------------------------------------------------------
__global__ __launch_bounds__(256) void attn_kernel(
    const _Float16* __restrict__ Qh, const _Float16* __restrict__ Kh,
    const _Float16* __restrict__ Vh, const unsigned char* __restrict__ mask,
    float* __restrict__ out)
{
  __shared__ _Float16 Pl[8][16 * 64];  // per-wave P staging (2 KB each)

  const int lane = threadIdx.x & 31;
  const int wid  = threadIdx.x >> 5;
  const int wave = (blockIdx.x * blockDim.x + threadIdx.x) >> 5;  // 0..1023
  const int bIdx = wave >> 7;
  const int t0   = (wave & 127) << 4;
  const int nl   = lane & 15;
  const int hl   = lane >> 4;
  const int kb8  = hl * 8;
  const int kb16 = hl * 16;

  const _Float16* Qb = Qh + (size_t)bIdx * Tn * Hn;
  const _Float16* Kb = Kh + (size_t)bIdx * Tn * Hn;
  const _Float16* Vb = Vh + (size_t)bIdx * Tn * Hn;
  const unsigned char* Mb = mask + (size_t)bIdx * Tn * Tn;

  // Q A-fragments (16x32 each) for head dims [0,32) and [32,64); loaded once.
  v16h aq0, aq1;
  {
    const _Float16* qr = Qb + (size_t)(t0 + nl) * Hn;
#pragma unroll
    for (int j = 0; j < 8; ++j) {
      aq0[j]     = qr[kb16 + j];
      aq0[8 + j] = qr[8 + kb16 + j];
      aq1[j]     = qr[32 + kb16 + j];
      aq1[8 + j] = qr[40 + kb16 + j];
    }
  }

  v8f acc[4] = {};          // O accumulator, 16x64
  float mrun[8], lrun[8];   // per-row running max / sum
#pragma unroll
  for (int r = 0; r < 8; ++r) { mrun[r] = -1e30f; lrun[r] = 0.0f; }

  _Float16* P = &Pl[wid][0];

  for (int kt = 0; kt < t0 + 16; kt += 64) {  // causal bound; keys <= 2047
    // Prefetch this block's V rows and next block's K rows (one row / lane).
    __builtin_prefetch(Vb + (size_t)(kt + lane) * Hn, 0, 0);
    __builtin_prefetch(Vb + (size_t)(kt + 32 + lane) * Hn, 0, 0);
    if (kt + 64 < t0 + 16) {
      __builtin_prefetch(Kb + (size_t)(kt + 64 + lane) * Hn, 0, 0);
      __builtin_prefetch(Kb + (size_t)(kt + 96 + lane) * Hn, 0, 0);
    }

    // S tiles: 4 x (16 queries x 16 keys), K-dim = 64 head dims.
    v8f S[4];
#pragma unroll
    for (int s = 0; s < 4; ++s) {
      const _Float16* kr = Kb + (size_t)(kt + 16 * s + nl) * Hn;
      const v16h b0 = *(const v16h*)(kr + kb16);
      const v16h b1 = *(const v16h*)(kr + 32 + kb16);
      v8f t = {};
      t = WMMA_F16(aq0, b0, t);
      t = WMMA_F16(aq1, b1, t);
      S[s] = t;
    }

    // Online softmax over 64 keys at once (scale = sqrt(H) = 8).
    float sc[8];
#pragma unroll
    for (int r = 0; r < 8; ++r) {
      const int qi = t0 + r + 8 * hl;
      const unsigned char* mr = Mb + (size_t)qi * Tn;
      float v[4];
#pragma unroll
      for (int s = 0; s < 4; ++s) {
        const int j = kt + 16 * s + nl;
        v[s] = S[s][r] * 8.0f;
        if (j > qi || mr[j]) v[s] = -1e30f;
      }
      float mx = fmaxf(fmaxf(v[0], v[1]), fmaxf(v[2], v[3]));
#pragma unroll
      for (int off = 1; off < 16; off <<= 1)
        mx = fmaxf(mx, __shfl_xor(mx, off, 32));
      const float mnew  = fmaxf(mrun[r], mx);
      const float scale = __expf(mrun[r] - mnew);
      float rs = 0.0f;
#pragma unroll
      for (int s = 0; s < 4; ++s) {
        const float e = __expf(v[s] - mnew);
        rs += e;
        P[(r + 8 * hl) * 64 + 16 * s + nl] = (_Float16)e;  // stream to LDS
      }
#pragma unroll
      for (int off = 1; off < 16; off <<= 1)
        rs += __shfl_xor(rs, off, 32);
      lrun[r] = lrun[r] * scale + rs;
      mrun[r] = mnew;
      sc[r] = scale;
    }
#pragma unroll
    for (int nt = 0; nt < 4; ++nt)
#pragma unroll
      for (int r = 0; r < 8; ++r) acc[nt][r] *= sc[r];

    asm volatile("s_wait_dscnt 0" ::: "memory");

    // P A-fragments (16x32 each) for key halves [kt,kt+32) and [kt+32,kt+64).
    v16h ap0, ap1;
    {
      const _Float16* pr = P + nl * 64;
#pragma unroll
      for (int j = 0; j < 8; ++j) {
        ap0[j]     = pr[kb16 + j];
        ap0[8 + j] = pr[8 + kb16 + j];
        ap1[j]     = pr[32 + kb16 + j];
        ap1[8 + j] = pr[40 + kb16 + j];
      }
    }

    // O += P (16x64) x V (64x64): 4 head-column tiles, 2 K-steps each.
#pragma unroll
    for (int nt = 0; nt < 4; ++nt) {
      v16h bv0, bv1;
#pragma unroll
      for (int j = 0; j < 16; ++j) {
        bv0[j] = Vb[(size_t)(kt + kb16 + j) * Hn + nt * 16 + nl];
        bv1[j] = Vb[(size_t)(kt + 32 + kb16 + j) * Hn + nt * 16 + nl];
      }
      acc[nt] = WMMA_F16(ap0, bv0, acc[nt]);
      acc[nt] = WMMA_F16(ap1, bv1, acc[nt]);
    }
  }

  // Epilogue: normalize and store fp32 output [B,T,64].
#pragma unroll
  for (int nt = 0; nt < 4; ++nt) {
#pragma unroll
    for (int r = 0; r < 8; ++r) {
      const float inv = 1.0f / lrun[r];
      out[((size_t)bIdx * Tn + t0 + r + 8 * hl) * Hn + nt * 16 + nl] =
          acc[nt][r] * inv;
    }
  }
}

// ---------------------------------------------------------------------------
extern "C" void kernel_launch(void* const* d_in, const int* in_sizes, int n_in,
                              void* d_out, int out_size, void* d_ws, size_t ws_size,
                              hipStream_t stream) {
  const float* x  = (const float*)d_in[0];
  const float* Wq = (const float*)d_in[1];
  const float* bq = (const float*)d_in[2];
  const float* Wk = (const float*)d_in[3];
  const float* bk = (const float*)d_in[4];
  const float* Wv = (const float*)d_in[5];
  const float* bv = (const float*)d_in[6];
  const unsigned char* mask = (const unsigned char*)d_in[7];
  float* out = (float*)d_out;

  const size_t qkvElems = (size_t)Bn * Tn * Hn;   // 1 M f16 each
  const size_t wElems   = (size_t)Hn * En;        // 64 K f16 each

  _Float16* Qh  = (_Float16*)d_ws;                // 2 MB
  _Float16* Kh  = Qh + qkvElems;                  // 2 MB
  _Float16* Vh  = Kh + qkvElems;                  // 2 MB
  _Float16* Whq = Vh + qkvElems;                  // 128 KB
  _Float16* Whk = Whq + wElems;                   // 128 KB
  _Float16* Whv = Whk + wElems;                   // 128 KB

  const dim3 block(256);

  const int n4 = (int)(wElems / 4);
  const dim3 cgrid((n4 + 255) / 256);
  cvt_f32_f16_kernel<<<cgrid, block, 0, stream>>>(Wq, Whq, n4);
  cvt_f32_f16_kernel<<<cgrid, block, 0, stream>>>(Wk, Whk, n4);
  cvt_f32_f16_kernel<<<cgrid, block, 0, stream>>>(Wv, Whv, n4);

  const dim3 grid((Bn * Tn / 16) / 8);            // 1024 waves total
  qkv_kernel<<<grid, block, 0, stream>>>(x, Whq, bq, Whk, bk, Whv, bv, Qh, Kh, Vh);
  attn_kernel<<<grid, block, 0, stream>>>(Qh, Kh, Vh, mask, out);
}
